// GNN_4157528343204
// MI455X (gfx1250) — compile-verified, matrix operand order checked by reading, not run on hardware
//
#include <hip/hip_runtime.h>

// ---------------------------------------------------------------------------
// GIN (5-layer) forward for MI455X / gfx1250, wave32 WMMA bf16 (f32 accum).
// Hot GEMM loop: 4x ds_load_b128 (A frags) + 2x global_load_b128 (packed B
// frag) + 2x v_wmma_f32_16x16x32_bf16 per K-step; no branches, no per-elem
// waits. Weights pre-packed to WMMA B-fragment layout (fits in L2).
// ---------------------------------------------------------------------------

typedef __attribute__((ext_vector_type(16))) __bf16 v16bf;
typedef __attribute__((ext_vector_type(8)))  __bf16 v8bf;
typedef __attribute__((ext_vector_type(2)))  __bf16 v2bf;
typedef __attribute__((ext_vector_type(8)))  float  v8f;
typedef __attribute__((ext_vector_type(2)))  float  v2f;
typedef __attribute__((ext_vector_type(4)))  unsigned int v4u;

#define D_IN    300
#define D_HID   600
#define KPAD1   320   // 300 -> mult of 32
#define KPAD2   608   // 600 -> mult of 32 (also hid row stride)
#define KB1     10    // KPAD1/32
#define KB2     19    // KPAD2/32
#define NT1     40    // col tiles packed for gemm1 (grid 10 blocks x 64 cols)
#define NT2     20    // col tiles packed for gemm2 (grid 5 blocks x 64 cols)

// K index of element e (0..15) of a v16bf fragment for lane-half (lane>>4),
// per CDNA5 ISA 7.12.2 "16-bit A-Matrix 16x32" (B uses same K pattern, lane
// low 4 bits select N). Elements 0..7 -> K = half*8 + e (contiguous),
// elements 8..15 -> K = 16 + half*8 + (e-8) (contiguous).
__device__ __forceinline__ int wmma_k_of_elem(int e, int half) {
    int j = e >> 1, p = e & 1;
    return ((j >= 4) ? 16 : 0) + half * 8 + ((j & 3) << 1) + p;
}

__device__ __forceinline__ v16bf frag_combine(v8bf lo, v8bf hi) {
    return __builtin_shufflevector(lo, hi, 0, 1, 2, 3, 4, 5, 6, 7,
                                           8, 9, 10, 11, 12, 13, 14, 15);
}

// -------------------------------- embedding --------------------------------
__global__ __launch_bounds__(256)
void embed_kernel(const int* __restrict__ x,
                  const float* __restrict__ xe1,   // 121 x 300
                  const float* __restrict__ xe2,   // 11 x 300
                  float* __restrict__ h, int N) {
    int tid = blockIdx.x * 256 + threadIdx.x;      // over N*300
    if (tid >= N * D_IN) return;
    int i = tid / D_IN, d = tid - i * D_IN;
    h[tid] = xe1[(size_t)x[2 * i] * D_IN + d] + xe2[(size_t)x[2 * i + 1] * D_IN + d];
}

// ---------------------- agg init (folds the self loop) ---------------------
__global__ __launch_bounds__(256)
void init_agg_kernel(const float* __restrict__ h,
                     const float* __restrict__ ee1l,  // edge_emb1[l] : 7 x 300
                     const float* __restrict__ ee2l,  // edge_emb2[l] : 3 x 300
                     float* __restrict__ agg, int N) {
    int tid = blockIdx.x * 256 + threadIdx.x;      // over N*300
    if (tid >= N * D_IN) return;
    int d = tid % D_IN;
    agg[tid] = h[tid] + ee1l[4 * D_IN + d] + ee2l[d];   // self loop attr [4,0]
}

// ------------------------- message + scatter-add ---------------------------
__global__ __launch_bounds__(256)
void scatter_kernel(const float* __restrict__ h,
                    const int* __restrict__ src, const int* __restrict__ dst,
                    const int* __restrict__ ea,      // E x 2
                    const float* __restrict__ ee1l,
                    const float* __restrict__ ee2l,
                    float* __restrict__ agg, int E) {
    int tid = blockIdx.x * 256 + threadIdx.x;      // over E*300
    if (tid >= E * D_IN) return;
    int e = tid / D_IN, d = tid - e * D_IN;
    int s  = src[e];
    int t  = dst[e];
    int a0 = ea[2 * e];
    int a1 = ea[2 * e + 1];
    float msg = h[(size_t)s * D_IN + d] + ee1l[a0 * D_IN + d] + ee2l[a1 * D_IN + d];
    __hip_atomic_fetch_add(&agg[(size_t)t * D_IN + d], msg,
                           __ATOMIC_RELAXED, __HIP_MEMORY_SCOPE_AGENT);
}

// ------------------- weight pre-pack to B-fragment layout ------------------
// out[((nt*kblocks + kb)*32 + lane)*16 + e] = bf16(W[k, n]) or 0 (pad).
__global__ __launch_bounds__(256)
void pack_w_kernel(const float* __restrict__ W, __bf16* __restrict__ out,
                   int K, int Ncols, int kblocks, int total) {
    int idx = blockIdx.x * 256 + threadIdx.x;
    if (idx >= total) return;
    int e    = idx & 15;
    int lane = (idx >> 4) & 31;
    int fb   = idx >> 9;            // fragment block = nt*kblocks + kb
    int kb   = fb % kblocks;
    int nt   = fb / kblocks;
    int k = kb * 32 + wmma_k_of_elem(e, lane >> 4);
    int n = nt * 16 + (lane & 15);
    float v = (k < K && n < Ncols) ? W[(size_t)k * Ncols + n] : 0.0f;
    out[idx] = (__bf16)v;
}

// ------------------- GEMM1: hid = relu(agg @ W1 + b1) ----------------------
// A: f32 N x 300 -> LDS bf16 tile 32 x 320. Out: bf16, row stride KPAD2=608,
// pad cols [600,608) zeroed so gemm2 staging is a straight vector copy.
__global__ __launch_bounds__(128)
void gemm1_kernel(const float* __restrict__ A,
                  const __bf16* __restrict__ Wpk,   // NT1 x KB1 x 32 x 16
                  const float* __restrict__ bias,   // 600
                  __bf16* __restrict__ out, int N) {
    __shared__ __bf16 Atile[32 * KPAD1];
    const int row0 = blockIdx.y * 32;
    const int tid  = threadIdx.x;

    // stage A (f32 -> bf16), vectorized as pairs; zero-pad K and OOB rows
    for (int idx = tid; idx < 32 * (KPAD1 / 2); idx += 128) {
        int m = idx / (KPAD1 / 2);
        int k = (idx - m * (KPAD1 / 2)) * 2;
        int row = row0 + m;
        v2f v = {0.0f, 0.0f};
        if (row < N && k < D_IN) v = *(const v2f*)(&A[(size_t)row * D_IN + k]);
        v2bf p; p[0] = (__bf16)v[0]; p[1] = (__bf16)v[1];
        *(v2bf*)(&Atile[m * KPAD1 + k]) = p;
    }
    __syncthreads();

    const int wave = tid >> 5;
    const int lane = tid & 31;
    const int half = lane >> 4;
    const int l16  = lane & 15;
    const int nt   = blockIdx.x * 4 + wave;          // 0..39, all packed
    const int n    = nt * 16 + l16;

    const __bf16* bp = Wpk + (((size_t)nt * KB1) * 32 + lane) * 16;

    v8f acc0 = {}, acc1 = {};
    for (int kbi = 0; kbi < KB1; ++kbi) {
        int kb = kbi * 32;
        v16bf b = *(const v16bf*)(bp + (size_t)kbi * 512);
        v8bf a0l = *(const v8bf*)(&Atile[l16 * KPAD1 + kb + half * 8]);
        v8bf a0h = *(const v8bf*)(&Atile[l16 * KPAD1 + kb + 16 + half * 8]);
        v8bf a1l = *(const v8bf*)(&Atile[(16 + l16) * KPAD1 + kb + half * 8]);
        v8bf a1h = *(const v8bf*)(&Atile[(16 + l16) * KPAD1 + kb + 16 + half * 8]);
        acc0 = __builtin_amdgcn_wmma_f32_16x16x32_bf16(
            false, frag_combine(a0l, a0h), false, b, (short)0, acc0, false, false);
        acc1 = __builtin_amdgcn_wmma_f32_16x16x32_bf16(
            false, frag_combine(a1l, a1h), false, b, (short)0, acc1, false, false);
    }

    if (n < KPAD2) {                                  // cols [600,608) store 0
        float bv = (n < D_HID) ? bias[n] : 0.0f;
#pragma unroll
        for (int r = 0; r < 8; ++r) {
            int m0 = row0 + r + 8 * half;
            int m1 = m0 + 16;
            float v0 = acc0[r] + bv, v1 = acc1[r] + bv;
            v0 = v0 > 0.0f ? v0 : 0.0f;
            v1 = v1 > 0.0f ? v1 : 0.0f;
            if (n >= D_HID) { v0 = 0.0f; v1 = 0.0f; }
            if (m0 < N) out[(size_t)m0 * KPAD2 + n] = (__bf16)v0;
            if (m1 < N) out[(size_t)m1 * KPAD2 + n] = (__bf16)v1;
        }
    }
}

// --------------------- GEMM2: h2 = hid @ W2 + b2 ---------------------------
// A: bf16 N x 608 (padded) -> LDS via 128-bit copies. Out: f32 N x 300.
__global__ __launch_bounds__(128)
void gemm2_kernel(const __bf16* __restrict__ A,
                  const __bf16* __restrict__ Wpk,   // NT2 x KB2 x 32 x 16
                  const float* __restrict__ bias,   // 300
                  float* __restrict__ out, int N) {
    __shared__ __bf16 Atile[32 * KPAD2];
    const int row0 = blockIdx.y * 32;
    const int tid  = threadIdx.x;

    // 32 rows x (1216/16 = 76) uint4 chunks
    for (int idx = tid; idx < 32 * 76; idx += 128) {
        int m = idx / 76, c = idx - m * 76;
        int row = row0 + m;
        v4u v = {0u, 0u, 0u, 0u};
        if (row < N) v = ((const v4u*)(A + (size_t)row * KPAD2))[c];
        ((v4u*)(&Atile[(size_t)m * KPAD2]))[c] = v;
    }
    __syncthreads();

    const int wave = tid >> 5;
    const int lane = tid & 31;
    const int half = lane >> 4;
    const int l16  = lane & 15;
    const int nt   = blockIdx.x * 4 + wave;          // 0..19, all packed
    const int n    = nt * 16 + l16;

    const __bf16* bp = Wpk + (((size_t)nt * KB2) * 32 + lane) * 16;

    v8f acc0 = {}, acc1 = {};
    for (int kbi = 0; kbi < KB2; ++kbi) {
        int kb = kbi * 32;
        v16bf b = *(const v16bf*)(bp + (size_t)kbi * 512);
        v8bf a0l = *(const v8bf*)(&Atile[l16 * KPAD2 + kb + half * 8]);
        v8bf a0h = *(const v8bf*)(&Atile[l16 * KPAD2 + kb + 16 + half * 8]);
        v8bf a1l = *(const v8bf*)(&Atile[(16 + l16) * KPAD2 + kb + half * 8]);
        v8bf a1h = *(const v8bf*)(&Atile[(16 + l16) * KPAD2 + kb + 16 + half * 8]);
        acc0 = __builtin_amdgcn_wmma_f32_16x16x32_bf16(
            false, frag_combine(a0l, a0h), false, b, (short)0, acc0, false, false);
        acc1 = __builtin_amdgcn_wmma_f32_16x16x32_bf16(
            false, frag_combine(a1l, a1h), false, b, (short)0, acc1, false, false);
    }

    if (n < D_IN) {
        float bv = bias[n];
#pragma unroll
        for (int r = 0; r < 8; ++r) {
            int m0 = row0 + r + 8 * half;
            int m1 = m0 + 16;
            if (m0 < N) out[(size_t)m0 * D_IN + n] = acc0[r] + bv;
            if (m1 < N) out[(size_t)m1 * D_IN + n] = acc1[r] + bv;
        }
    }
}

// ------------------------------ batch-norm ---------------------------------
__global__ __launch_bounds__(256)
void zero_stats_kernel(float* __restrict__ stats) {
    int tid = blockIdx.x * 256 + threadIdx.x;
    if (tid < 2 * D_IN) stats[tid] = 0.0f;
}

__global__ __launch_bounds__(256)
void bn_stats_kernel(const float* __restrict__ h2,
                     float* __restrict__ stats, int N) {
    const int row0 = blockIdx.x * 64;
    const int tid  = threadIdx.x;
    const int c0 = tid, c1 = tid + 256;
    float s0 = 0.f, q0 = 0.f, s1 = 0.f, q1 = 0.f;
    for (int r = 0; r < 64; ++r) {
        int row = row0 + r;
        if (row < N) {
            const float* p = h2 + (size_t)row * D_IN;
            if (c0 < D_IN) { float v = p[c0]; s0 += v; q0 += v * v; }
            if (c1 < D_IN) { float v = p[c1]; s1 += v; q1 += v * v; }
        }
    }
    if (c0 < D_IN) {
        __hip_atomic_fetch_add(&stats[c0],        s0, __ATOMIC_RELAXED, __HIP_MEMORY_SCOPE_AGENT);
        __hip_atomic_fetch_add(&stats[D_IN + c0], q0, __ATOMIC_RELAXED, __HIP_MEMORY_SCOPE_AGENT);
    }
    if (c1 < D_IN) {
        __hip_atomic_fetch_add(&stats[c1],        s1, __ATOMIC_RELAXED, __HIP_MEMORY_SCOPE_AGENT);
        __hip_atomic_fetch_add(&stats[D_IN + c1], q1, __ATOMIC_RELAXED, __HIP_MEMORY_SCOPE_AGENT);
    }
}

__global__ __launch_bounds__(256)
void bn_norm_kernel(const float* __restrict__ h2,
                    const float* __restrict__ stats,
                    const float* __restrict__ gamma,
                    const float* __restrict__ beta,
                    float* __restrict__ out, int N, int do_relu) {
    int tid = blockIdx.x * 256 + threadIdx.x;      // over N*300
    if (tid >= N * D_IN) return;
    int c = tid % D_IN;
    float invN  = 1.0f / (float)N;
    float mean  = stats[c] * invN;
    float var   = stats[D_IN + c] * invN - mean * mean;
    float scale = rsqrtf(var + 1e-5f) * gamma[c];
    float v = (h2[tid] - mean) * scale + beta[c];
    if (do_relu) v = v > 0.0f ? v : 0.0f;
    out[tid] = v;
}

// ------------------------------ host driver --------------------------------
extern "C" void kernel_launch(void* const* d_in, const int* in_sizes, int n_in,
                              void* d_out, int out_size, void* d_ws, size_t ws_size,
                              hipStream_t stream) {
    const int N = in_sizes[0] / 2;
    const int E = in_sizes[1] / 2;

    const int*   x     = (const int*)d_in[0];
    const int*   eidx  = (const int*)d_in[1];   // [2, E]: src then dst
    const int*   eattr = (const int*)d_in[2];   // [E, 2]
    const float* xe1   = (const float*)d_in[3];
    const float* xe2   = (const float*)d_in[4];
    const float* ee1   = (const float*)d_in[5]; // [5, 7, 300]
    const float* ee2   = (const float*)d_in[6]; // [5, 3, 300]
    const float* w1    = (const float*)d_in[7]; // [5, 300, 600]
    const float* b1    = (const float*)d_in[8]; // [5, 600]
    const float* w2    = (const float*)d_in[9]; // [5, 600, 300]
    const float* b2    = (const float*)d_in[10];// [5, 300]
    const float* gamma = (const float*)d_in[11];
    const float* beta  = (const float*)d_in[12];

    // workspace layout
    char* ws = (char*)d_ws;
    float*  buf_h   = (float*)(ws);                          // N x 300 f32
    float*  buf_agg = (float*)(ws + (size_t)N * 1200);       // N x 300 f32 (also h2)
    __bf16* buf_hid = (__bf16*)(ws + (size_t)N * 2400);      // N x 608 bf16
    size_t  owp1    = (size_t)N * 2400 + (size_t)N * 1216;   // 16B aligned
    const int tot1  = NT1 * KB1 * 512;                        // packed elems
    const int tot2  = NT2 * KB2 * 512;
    __bf16* wpk1    = (__bf16*)(ws + owp1);
    __bf16* wpk2    = (__bf16*)(ws + owp1 + (size_t)tot1 * 2);
    float*  stats   = (float*)(ws + owp1 + (size_t)(tot1 + tot2) * 2);

    const int nd  = N * D_IN;
    const int ed  = E * D_IN;
    const int gnd = (nd + 255) / 256;

    embed_kernel<<<gnd, 256, 0, stream>>>(x, xe1, xe2, buf_h, N);

    const int mtiles = (N + 31) / 32;
    for (int l = 0; l < 5; ++l) {
        const float* ee1l = ee1 + (size_t)l * 7 * D_IN;
        const float* ee2l = ee2 + (size_t)l * 3 * D_IN;

        init_agg_kernel<<<gnd, 256, 0, stream>>>(buf_h, ee1l, ee2l, buf_agg, N);
        scatter_kernel<<<(ed + 255) / 256, 256, 0, stream>>>(
            buf_h, eidx, eidx + E, eattr, ee1l, ee2l, buf_agg, E);

        pack_w_kernel<<<(tot1 + 255) / 256, 256, 0, stream>>>(
            w1 + (size_t)l * D_IN * D_HID, wpk1, D_IN, D_HID, KB1, tot1);
        pack_w_kernel<<<(tot2 + 255) / 256, 256, 0, stream>>>(
            w2 + (size_t)l * D_HID * D_IN, wpk2, D_HID, D_IN, KB2, tot2);

        gemm1_kernel<<<dim3(NT1 / 4, mtiles), 128, 0, stream>>>(
            buf_agg, wpk1, b1 + (size_t)l * D_HID, buf_hid, N);
        gemm2_kernel<<<dim3(NT2 / 4, mtiles), 128, 0, stream>>>(
            buf_hid, wpk2, b2 + (size_t)l * D_IN, buf_agg, N);   // h2 -> buf_agg

        zero_stats_kernel<<<3, 256, 0, stream>>>(stats);
        bn_stats_kernel<<<(N + 63) / 64, 256, 0, stream>>>(buf_agg, stats, N);

        float* outp = (l == 4) ? (float*)d_out : buf_h;
        bn_norm_kernel<<<gnd, 256, 0, stream>>>(
            buf_agg, stats, gamma + (size_t)l * D_IN, beta + (size_t)l * D_IN,
            outp, N, (l < 4) ? 1 : 0);
    }
}
